// decoder_16707422781790
// MI455X (gfx1250) — compile-verified, matrix operand order checked by reading, not run on hardware
//
#include <hip/hip_runtime.h>

typedef unsigned short u16;
typedef __attribute__((ext_vector_type(16))) __bf16        v16bf;
typedef __attribute__((ext_vector_type(8)))  float         v8f;
typedef __attribute__((ext_vector_type(8)))  unsigned int  v8u;

union Frag { v8u u; v16bf b; };

// Problem constants
#define BB   128
#define TT   64
#define ROWS (BB * TT)   // 8192
#define VV   8000
#define HH   600         // H1 = H2 = E = C = 600
#define HP   640         // padded H (multiple of 64)
#define NG   (3 * HP)    // 1920 = padded 3H gate width
#define KX   1216        // padded E+C (1200 -> 1216, multiple of 64)

__device__ __forceinline__ u16 f2bf(float f) {
  unsigned u = __float_as_uint(f);
  unsigned r = 0x7FFFu + ((u >> 16) & 1u);   // round-to-nearest-even
  return (u16)((u + r) >> 16);
}

__device__ __forceinline__ void load_frags(const unsigned* ap0, const unsigned* ap1,
                                           const unsigned* bp0, const unsigned* bp1,
                                           const unsigned* bp2, const unsigned* bp3,
                                           int pb, Frag (&a)[2], Frag (&b)[4]) {
#pragma unroll
  for (int v = 0; v < 8; ++v) {
    const int p = pb + v + (v & 4);          // dwords {0..3,8..11} + 4h
    a[0].u[v] = ap0[p]; a[1].u[v] = ap1[p];
    b[0].u[v] = bp0[p]; b[1].u[v] = bp1[p];
    b[2].u[v] = bp2[p]; b[3].u[v] = bp3[p];
  }
}

__device__ __forceinline__ void wmma8(const Frag (&a)[2], const Frag (&b)[4],
                                      v8f (&acc)[8]) {
#pragma unroll
  for (int mi = 0; mi < 2; ++mi)
#pragma unroll
    for (int nj = 0; nj < 4; ++nj)
      acc[mi * 4 + nj] = __builtin_amdgcn_wmma_f32_16x16x32_bf16(
          false, a[mi].b, false, b[nj].b, (short)0, acc[mi * 4 + nj],
          false, false);
}

// ---------------------------------------------------------------------------
// Core WMMA GEMM: C[M,N] = A[M,K] * B[K,N] (+bias), A row-major bf16,
// B supplied TRANSPOSED row-major bf16 (BT[N][K]); both fragments use the
// same packed-pair gather matching the CDNA5 16-bit A-matrix VGPR layout
// (dwords {0..3,8..11}+4h -> coalesced by the compiler into 2x b128).
// One wave computes a 32x64 tile: 2 A-frags x 4 B-frags -> 8 accumulators.
// K loop advances 64/iter with two fixed ping-pong buffer sets (no copies),
// so fragment loads for one phase overlap the other phase's 8 wmmas.
// Dims must be multiples of 32 (M), 64 (N), 64 (K).  EXEC stays all-1s.
// ---------------------------------------------------------------------------
__global__ __launch_bounds__(256)
void k_gemm_bf16(const u16* __restrict__ A, const u16* __restrict__ BT,
                 const float* __restrict__ bias, float* __restrict__ C,
                 int M, int N, int K, int lda, int ldbt, int ldc) {
  const int lane = threadIdx.x & 31;
  const int wave = threadIdx.x >> 5;
  const int tiles_n = N >> 6;
  const int tile = blockIdx.x * 8 + wave;
  if (tile >= (M >> 5) * tiles_n) return;      // wave-uniform -> EXEC all-1s
  const int m0 = (tile / tiles_n) << 5;
  const int n0 = (tile % tiles_n) << 6;
  const int h  = lane >> 4;                    // which K-half this lane holds
  const int r  = lane & 15;                    // row (A) / col (B/C) index

  const unsigned* ap0 = reinterpret_cast<const unsigned*>(A + (size_t)(m0 +  0 + r) * lda);
  const unsigned* ap1 = reinterpret_cast<const unsigned*>(A + (size_t)(m0 + 16 + r) * lda);
  const unsigned* bp0 = reinterpret_cast<const unsigned*>(BT + (size_t)(n0 +  0 + r) * ldbt);
  const unsigned* bp1 = reinterpret_cast<const unsigned*>(BT + (size_t)(n0 + 16 + r) * ldbt);
  const unsigned* bp2 = reinterpret_cast<const unsigned*>(BT + (size_t)(n0 + 32 + r) * ldbt);
  const unsigned* bp3 = reinterpret_cast<const unsigned*>(BT + (size_t)(n0 + 48 + r) * ldbt);

  v8f acc[8];
#pragma unroll
  for (int i = 0; i < 8; ++i) acc[i] = (v8f){};

  Frag a0[2], b0[4], a1[2], b1[4];
  load_frags(ap0, ap1, bp0, bp1, bp2, bp3, 4 * h, a0, b0);

  for (int k0 = 0; k0 < K; k0 += 64) {
    load_frags(ap0, ap1, bp0, bp1, bp2, bp3, ((k0 + 32) >> 1) + 4 * h, a1, b1);
    wmma8(a0, b0, acc);
    if (k0 + 64 < K)
      load_frags(ap0, ap1, bp0, bp1, bp2, bp3, ((k0 + 64) >> 1) + 4 * h, a0, b0);
    wmma8(a1, b1, acc);
  }

  float bi[4] = {0.f, 0.f, 0.f, 0.f};
  if (bias) {
#pragma unroll
    for (int nj = 0; nj < 4; ++nj) bi[nj] = bias[n0 + nj * 16 + r];
  }
#pragma unroll
  for (int mi = 0; mi < 2; ++mi) {
#pragma unroll
    for (int i = 0; i < 8; ++i) {
      // C layout: VGPR i holds M = (m0+16mi) + i + 8h, N = n0 + 16nj + (lane&15)
      float* c = C + (size_t)(m0 + mi * 16 + i + 8 * h) * ldc + n0 + r;
#pragma unroll
      for (int nj = 0; nj < 4; ++nj)
        c[nj * 16] = acc[mi * 4 + nj][i] + bi[nj];
    }
  }
}

// --------------------------- token extraction -------------------------------
__global__ __launch_bounds__(256)
void k_tok(const float* __restrict__ Y, int* __restrict__ tok) {
  __shared__ int s;
  const int row = blockIdx.x;
  if (threadIdx.x == 0) s = 0;
  __syncthreads();
  const float* y = Y + (size_t)row * VV;
  for (int i = threadIdx.x; i < VV; i += 256)
    if (y[i] > 0.5f) s = i;                    // one-hot: single writer
  __syncthreads();
  if (threadIdx.x == 0) tok[row] = s;
}

// --------- build x = [emb(tok)+b_emb | context | 0-pad] as bf16 --------------
__global__ __launch_bounds__(256)
void k_build_x(const int* __restrict__ tok, const float* __restrict__ W_emb,
               const float* __restrict__ b_emb, const float* __restrict__ ctx,
               u16* __restrict__ x) {
  const int row = blockIdx.x;
  const int tk  = tok[row];
  u16* xr = x + (size_t)row * KX;
  const float* er = W_emb + (size_t)tk * HH;
  const float* cr = ctx   + (size_t)row * HH;
  for (int c = threadIdx.x; c < KX; c += 256) {
    float v = 0.f;
    if (c < HH)          v = er[c] + b_emb[c];
    else if (c < 2 * HH) v = cr[c - HH];
    xr[c] = f2bf(v);
  }
}

// --- weight convert: f32 W[K][N] -> bf16 WT[Np][Kp] (transpose+pad+gate-remap)
__global__ __launch_bounds__(256)
void k_cvt_wt(const float* __restrict__ W, u16* __restrict__ WT,
              int K, int N, int Kp, int Np, int gsrc, int gdst) {
  const size_t idx = (size_t)blockIdx.x * 256 + threadIdx.x;
  if (idx >= (size_t)Np * Kp) return;
  const int np = (int)(idx / Kp);
  const int k  = (int)(idx % Kp);
  float val = 0.f;
  if (k < K) {
    if (gdst) {                               // gate remap: col g*gdst+i <- g*gsrc+i
      const int g = np / gdst, i = np % gdst;
      if (i < gsrc) val = W[(size_t)k * N + g * gsrc + i];
    } else if (np < N) {
      val = W[(size_t)k * N + np];
    }
  }
  WT[idx] = f2bf(val);
}

__global__ __launch_bounds__(256)
void k_pad_bias(const float* __restrict__ b, float* __restrict__ bp) {
  const int i = blockIdx.x * 256 + threadIdx.x;
  if (i >= NG) return;
  const int g = i / HP, j = i % HP;
  bp[i] = (j < HH) ? b[g * HH + j] : 0.f;
}

__global__ __launch_bounds__(256)
void k_zero_u16(u16* __restrict__ p, size_t n) {
  const size_t i = (size_t)blockIdx.x * 256 + threadIdx.x;
  if (i < n) p[i] = 0;
}

__global__ __launch_bounds__(256)
void k_init_state(const float* __restrict__ h0, float* __restrict__ hf,
                  u16* __restrict__ hb) {
  const int idx = blockIdx.x * 256 + threadIdx.x;
  if (idx >= BB * HP) return;
  const int b = idx / HP, i = idx % HP;
  const float v = (i < HH) ? h0[b * HH + i] : 0.f;
  if (i < HH) hf[b * HH + i] = v;
  hb[idx] = f2bf(v);
}

// ----------------------------- GRU gate update ------------------------------
__global__ __launch_bounds__(256)
void k_gru(const float* __restrict__ gx, const float* __restrict__ hW,
           float* __restrict__ hf, u16* __restrict__ hb,
           u16* __restrict__ hall, int t) {
  const int idx = blockIdx.x * 256 + threadIdx.x;
  if (idx >= BB * HH) return;
  const int b = idx / HH, i = idx % HH;
  const size_t grow = (size_t)(b * TT + t) * NG;
  const size_t wrow = (size_t)b * NG;
  const float r = 1.f / (1.f + expf(-(gx[grow + i]          + hW[wrow + i])));
  const float z = 1.f / (1.f + expf(-(gx[grow + HP + i]     + hW[wrow + HP + i])));
  const float n = tanhf(gx[grow + 2 * HP + i] + r * hW[wrow + 2 * HP + i]);
  const float hp = hf[b * HH + i];
  const float hn = (1.f - z) * n + z * hp;
  hf[b * HH + i] = hn;
  const u16 v = f2bf(hn);
  hb[b * HP + i] = v;
  hall[(size_t)(b * TT + t) * HP + i] = v;
}

// ------------------------- in-place row softmax -----------------------------
__global__ __launch_bounds__(256)
void k_softmax(float* __restrict__ X, int ncol) {
  __shared__ float red[256];
  float* x = X + (size_t)blockIdx.x * ncol;
  float m = -3.4e38f;
  for (int i = threadIdx.x; i < ncol; i += 256) m = fmaxf(m, x[i]);
  red[threadIdx.x] = m; __syncthreads();
  for (int s = 128; s > 0; s >>= 1) {
    if (threadIdx.x < s) red[threadIdx.x] = fmaxf(red[threadIdx.x], red[threadIdx.x + s]);
    __syncthreads();
  }
  m = red[0]; __syncthreads();
  float sum = 0.f;
  for (int i = threadIdx.x; i < ncol; i += 256) {
    const float e = expf(x[i] - m);
    x[i] = e; sum += e;
  }
  red[threadIdx.x] = sum; __syncthreads();
  for (int s = 128; s > 0; s >>= 1) {
    if (threadIdx.x < s) red[threadIdx.x] += red[threadIdx.x + s];
    __syncthreads();
  }
  const float inv = 1.f / red[0];
  for (int i = threadIdx.x; i < ncol; i += 256) x[i] *= inv;
}

// ============================================================================
extern "C" void kernel_launch(void* const* d_in, const int* in_sizes, int n_in,
                              void* d_out, int out_size, void* d_ws, size_t ws_size,
                              hipStream_t stream) {
  const float* Y       = (const float*)d_in[0];
  const float* context = (const float*)d_in[1];
  const float* h1_0    = (const float*)d_in[2];
  const float* h2_0    = (const float*)d_in[3];
  const float* W_emb   = (const float*)d_in[4];
  const float* b_emb   = (const float*)d_in[5];
  const float* W1x     = (const float*)d_in[6];
  const float* W1h     = (const float*)d_in[7];
  const float* b1      = (const float*)d_in[8];
  const float* W2x     = (const float*)d_in[9];
  const float* W2h     = (const float*)d_in[10];
  const float* b2      = (const float*)d_in[11];
  const float* Wout    = (const float*)d_in[12];
  const float* bout    = (const float*)d_in[13];
  float* out = (float*)d_out;

  char* ws = (char*)d_ws;
  size_t off = 0;
  auto alc = [&](size_t bytes) -> void* {
    void* p = (void*)(ws + off);
    off += (bytes + 255) & ~(size_t)255;
    return p;
  };
  u16*   x_b    = (u16*)  alc((size_t)ROWS * KX * 2);
  u16*   w1x_t  = (u16*)  alc((size_t)NG * KX * 2);
  u16*   w1h_t  = (u16*)  alc((size_t)NG * HP * 2);
  u16*   w2x_t  = (u16*)  alc((size_t)NG * HP * 2);
  u16*   w2h_t  = (u16*)  alc((size_t)NG * HP * 2);
  u16*   wout_t = (u16*)  alc((size_t)VV * HP * 2);
  float* b1p    = (float*)alc((size_t)NG * 4);
  float* b2p    = (float*)alc((size_t)NG * 4);
  float* gx     = (float*)alc((size_t)ROWS * NG * 4);   // reused for gx1 then gx2
  u16*   h1all  = (u16*)  alc((size_t)ROWS * HP * 2);
  u16*   h2all  = (u16*)  alc((size_t)ROWS * HP * 2);
  float* h1f    = (float*)alc((size_t)BB * HH * 4);
  float* h2f    = (float*)alc((size_t)BB * HH * 4);
  u16*   h1b    = (u16*)  alc((size_t)BB * HP * 2);
  u16*   h2b    = (u16*)  alc((size_t)BB * HP * 2);
  float* hWbuf  = (float*)alc((size_t)BB * NG * 4);
  int*   tok    = (int*)  alc((size_t)ROWS * 4);

  auto blk = [](size_t n) { return (int)((n + 255) / 256); };
  auto gemm = [&](const u16* A, const u16* BTm, const float* bias, float* C,
                  int M, int N, int K, int lda, int ldbt, int ldc) {
    const int tiles = (M / 32) * (N / 64);
    k_gemm_bf16<<<(tiles + 7) / 8, 256, 0, stream>>>(A, BTm, bias, C, M, N, K, lda, ldbt, ldc);
  };

  // 1) one-hot -> token ids; build x = [emb | ctx] in bf16
  k_tok<<<ROWS, 256, 0, stream>>>(Y, tok);
  k_build_x<<<ROWS, 256, 0, stream>>>(tok, W_emb, b_emb, context, x_b);

  // 2) weights -> transposed/padded bf16 (gate-remapped so gate g starts at g*HP)
  k_cvt_wt<<<blk((size_t)NG * KX), 256, 0, stream>>>(W1x, w1x_t, 2 * HH, 3 * HH, KX, NG, HH, HP);
  k_cvt_wt<<<blk((size_t)NG * HP), 256, 0, stream>>>(W1h, w1h_t, HH, 3 * HH, HP, NG, HH, HP);
  k_cvt_wt<<<blk((size_t)NG * HP), 256, 0, stream>>>(W2x, w2x_t, HH, 3 * HH, HP, NG, HH, HP);
  k_cvt_wt<<<blk((size_t)NG * HP), 256, 0, stream>>>(W2h, w2h_t, HH, 3 * HH, HP, NG, HH, HP);
  k_cvt_wt<<<blk((size_t)VV * HP), 256, 0, stream>>>(Wout, wout_t, HH, VV, HP, VV, 0, 0);
  k_pad_bias<<<blk(NG), 256, 0, stream>>>(b1, b1p);
  k_pad_bias<<<blk(NG), 256, 0, stream>>>(b2, b2p);

  // 3) init padded state + zero h_all pad columns
  k_zero_u16<<<blk((size_t)ROWS * HP), 256, 0, stream>>>(h1all, (size_t)ROWS * HP);
  k_zero_u16<<<blk((size_t)ROWS * HP), 256, 0, stream>>>(h2all, (size_t)ROWS * HP);
  k_init_state<<<blk(BB * HP), 256, 0, stream>>>(h1_0, h1f, h1b);
  k_init_state<<<blk(BB * HP), 256, 0, stream>>>(h2_0, h2f, h2b);

  // 4) gx1 = x @ W1x + b1   [8192 x 1920]
  gemm(x_b, w1x_t, b1p, gx, ROWS, NG, KX, KX, KX, NG);

  // 5) GRU layer 1: per-step WMMA h@W1h then gates
  for (int t = 0; t < TT; ++t) {
    gemm(h1b, w1h_t, nullptr, hWbuf, BB, NG, HP, HP, HP, NG);
    k_gru<<<blk(BB * HH), 256, 0, stream>>>(gx, hWbuf, h1f, h1b, h1all, t);
  }

  // 6) gx2 = h1 @ W2x + b2 (reuse gx buffer)
  gemm(h1all, w2x_t, b2p, gx, ROWS, NG, HP, HP, HP, NG);

  // 7) GRU layer 2
  for (int t = 0; t < TT; ++t) {
    gemm(h2b, w2h_t, nullptr, hWbuf, BB, NG, HP, HP, HP, NG);
    k_gru<<<blk(BB * HH), 256, 0, stream>>>(gx, hWbuf, h2f, h2b, h2all, t);
  }

  // 8) logits = h2 @ Wout + bout -> d_out (row = b*T+t matches [B,T,V]); softmax
  gemm(h2all, wout_t, bout, out, ROWS, VV, HP, HP, HP, VV);
  k_softmax<<<ROWS, 256, 0, stream>>>(out, VV);
}
